// DotProductAttention_2448131359415
// MI455X (gfx1250) — matrix operand order repaired and checked by reading, not compiled, not run on hardware
//
#include <hip/hip_runtime.h>

typedef __bf16 v16bf __attribute__((ext_vector_type(16)));
typedef float  v8f   __attribute__((ext_vector_type(8)));
typedef unsigned int u32x4 __attribute__((ext_vector_type(4)));
typedef int          i32x4 __attribute__((ext_vector_type(4)));
typedef int          i32x8 __attribute__((ext_vector_type(8)));

constexpr int NB  = 16;      // batch
constexpr int LQ  = 2048;
constexpr int LV  = 2048;
constexpr int DIM = 512;
constexpr int BM  = 128;     // q rows per workgroup (8 waves * 16)
constexpr int BN  = 32;      // v rows per tile (TDM tile = 512 x 32 fp32 = 64 KiB)
constexpr int PAD = 8;       // halves of padding (keeps 16B align, kills bank conflicts)
constexpr int NT  = LV / BN; // 64 tiles per sweep

union FragBF {
  v16bf v;
  uint4 q[2];
  unsigned short u[16];
};

__device__ __forceinline__ unsigned short f2bf(float f) {
  unsigned u = __float_as_uint(f);
  u += 0x7FFFu + ((u >> 16) & 1u);   // round-to-nearest-even
  return (unsigned short)(u >> 16);
}

// Issue a TDM load of one V tile (BN x DIM fp32, row-major) into LDS staging.
// D# built per CDNA5 ISA ch.8: group0 = {count, lds_addr, global_addr, type=2},
// group1 = {data_size=4B, tensor dims/strides, tile dims}. 2-D tensor -> groups 2/3 zero.
__device__ __forceinline__ void issue_tdm(const float* gsrc, unsigned lds_off) {
  const unsigned long long ga = (unsigned long long)(uintptr_t)gsrc;
  u32x4 g0;
  g0[0] = 1u;                                          // count=1 (valid), user mode
  g0[1] = lds_off;                                     // lds_addr (bytes)
  g0[2] = (unsigned)(ga & 0xFFFFFFFFu);                // global_addr[31:0]
  g0[3] = (unsigned)((ga >> 32) & 0x01FFFFFFu) | (2u << 30);  // global_addr[56:32] | type=2
  i32x8 g1;
  g1[0] = (int)(2u << 16);                             // workgroup_mask=0, data_size=2 (4 bytes)
  g1[1] = (int)((unsigned)(DIM & 0xFFFF) << 16);       // tensor_dim0[15:0]  (bits 63:48)
  g1[2] = (int)((unsigned)(LV  & 0xFFFF) << 16);       // tensor_dim0[31:16]=0 | tensor_dim1[15:0]
  g1[3] = (int)((unsigned)(DIM & 0xFFFF) << 16);       // tensor_dim1[31:16]=0 | tile_dim0=DIM
  g1[4] = (int)(BN & 0xFFFF);                          // tile_dim1=BN | tile_dim2=0
  g1[5] = (int)DIM;                                    // tensor_dim0_stride[31:0] (elements)
  g1[6] = 0;                                           // stride0[47:32] | tensor_dim1_stride lo
  g1[7] = 0;
  const i32x4 z4 = {0, 0, 0, 0};
#if defined(__clang_major__) && (__clang_major__ >= 23)
  const i32x8 z8 = {0, 0, 0, 0, 0, 0, 0, 0};
  __builtin_amdgcn_tensor_load_to_lds(g0, g1, z4, z4, z8, 0);
#else
  __builtin_amdgcn_tensor_load_to_lds(g0, g1, z4, z4, 0);
#endif
}

__global__ __launch_bounds__(256)
void dpa_fused_kernel(const float* __restrict__ Q, const float* __restrict__ V,
                      float* __restrict__ ctx, float* __restrict__ attn) {
  __shared__ float          sStage[2][BN][DIM];        // TDM double buffer (fp32)  128 KiB
  __shared__ unsigned short sV [BN ][DIM + PAD];       // V tile bf16 row-major      ~33 KiB
  __shared__ unsigned short sVt[DIM][BN  + PAD];       // V tile bf16 transposed     ~40 KiB
  __shared__ unsigned short sP [8][16][BN + PAD];      // per-wave P staging         ~10 KiB

  const int tid  = threadIdx.x;
  const int wave = tid >> 5;
  const int lane = tid & 31;
  const int grp  = lane >> 4;   // half-wave group (WMMA fragment layout)
  const int ln   = lane & 15;
  const int bb    = blockIdx.x >> 4;   // batch
  const int qbase = (blockIdx.x & 15) * BM;

  const float* vbatch = V + (size_t)bb * LV * DIM;

  // ---- load this wave's Q rows once, as bf16 A-fragments in registers ----
  FragBF aQ[16];
  {
    const float* qrow = Q + ((size_t)bb * LQ + (qbase + wave * 16 + ln)) * DIM;
    #pragma unroll
    for (int kc = 0; kc < 16; ++kc) {
      const int d0 = kc * 32 + grp * 8;
      const float4 f0 = *(const float4*)(qrow + d0);
      const float4 f1 = *(const float4*)(qrow + d0 + 4);
      const float4 f2 = *(const float4*)(qrow + d0 + 16);
      const float4 f3 = *(const float4*)(qrow + d0 + 20);
      aQ[kc].u[ 0]=f2bf(f0.x); aQ[kc].u[ 1]=f2bf(f0.y); aQ[kc].u[ 2]=f2bf(f0.z); aQ[kc].u[ 3]=f2bf(f0.w);
      aQ[kc].u[ 4]=f2bf(f1.x); aQ[kc].u[ 5]=f2bf(f1.y); aQ[kc].u[ 6]=f2bf(f1.z); aQ[kc].u[ 7]=f2bf(f1.w);
      aQ[kc].u[ 8]=f2bf(f2.x); aQ[kc].u[ 9]=f2bf(f2.y); aQ[kc].u[10]=f2bf(f2.z); aQ[kc].u[11]=f2bf(f2.w);
      aQ[kc].u[12]=f2bf(f3.x); aQ[kc].u[13]=f2bf(f3.y); aQ[kc].u[14]=f2bf(f3.z); aQ[kc].u[15]=f2bf(f3.w);
    }
  }

  // Convert fp32 staging tile -> bf16 sV (row-major) + sVt (transposed)
  auto convert_tile = [&](int buf) {
    for (int idx = tid; idx < BN * (DIM / 4); idx += 256) {
      const int vr = idx >> 7;          // DIM/4 = 128 float4 per row
      const int c4 = idx & 127;
      const float4 f = *(const float4*)(&sStage[buf][vr][c4 * 4]);
      const unsigned short h0=f2bf(f.x), h1=f2bf(f.y), h2=f2bf(f.z), h3=f2bf(f.w);
      uint2 pk;
      pk.x = (unsigned)h0 | ((unsigned)h1 << 16);
      pk.y = (unsigned)h2 | ((unsigned)h3 << 16);
      *reinterpret_cast<uint2*>(&sV[vr][c4 * 4]) = pk;
      const int d = c4 * 4;
      sVt[d + 0][vr] = h0; sVt[d + 1][vr] = h1;
      sVt[d + 2][vr] = h2; sVt[d + 3][vr] = h3;
    }
  };

  auto kick = [&](int tile, int buf) {
    issue_tdm(vbatch + (size_t)tile * BN * DIM,
              (unsigned)(uintptr_t)&sStage[buf][0][0]);
  };

  // S tile (16x16): rows = this wave's q rows, cols = nsub*16.. of current V tile
  auto computeS = [&](int nsub) {
    v8f c = {0.f,0.f,0.f,0.f,0.f,0.f,0.f,0.f};
    #pragma unroll
    for (int kc = 0; kc < 16; ++kc) {
      FragBF bV;
      const uint4* bp = reinterpret_cast<const uint4*>(&sV[nsub * 16 + ln][kc * 32 + grp * 8]);
      bV.q[0] = bp[0]; bV.q[1] = bp[2];
      c = __builtin_amdgcn_wmma_f32_16x16x32_bf16(false, aQ[kc].v, false, bV.v,
                                                  (short)0, c, false, false);
    }
    return c;
  };

  float mrun[8], lrun[8];
  #pragma unroll
  for (int i = 0; i < 8; ++i) { mrun[i] = -__builtin_inff(); lrun[i] = 0.f; }

  // prime the TDM pipeline with tile 0
  if (wave == 0) kick(0, 0);

  // ================= sweep 0: online softmax statistics (registers only) =================
  for (int vt = 0; vt < NT; ++vt) {
    if (wave == 0) __builtin_amdgcn_s_wait_tensorcnt((short)0);  // tile vt landed in staging
    __syncthreads();                                             // publish; prior compute done
    if (wave == 0) {
      const int nxt = (vt + 1 < NT) ? vt + 1 : 0;                // wraps into sweep 1's tile 0
      kick(nxt, (vt + 1) & 1);
    }
    convert_tile(vt & 1);
    __syncthreads();
    #pragma unroll
    for (int nsub = 0; nsub < BN / 16; ++nsub) {
      v8f c = computeS(nsub);
      #pragma unroll
      for (int v = 0; v < 8; ++v) {
        float x = c[v];
        float tm = x;
        #pragma unroll
        for (int m = 1; m < 16; m <<= 1) tm = fmaxf(tm, __shfl_xor(tm, m, 32));
        const float nm = fmaxf(mrun[v], tm);
        float ts = __expf(x - nm);
        #pragma unroll
        for (int m = 1; m < 16; m <<= 1) ts += __shfl_xor(ts, m, 32);
        lrun[v] = lrun[v] * __expf(mrun[v] - nm) + ts;
        mrun[v] = nm;
      }
    }
  }

  float linv[8];
  #pragma unroll
  for (int i = 0; i < 8; ++i) linv[i] = 1.0f / lrun[i];

  // ================= sweep 1: write attn, accumulate context = P @ V =================
  v8f cacc[32];  // 16 q-rows x full D=512 per wave, fp32
  {
    const v8f z = {0.f,0.f,0.f,0.f,0.f,0.f,0.f,0.f};
    #pragma unroll
    for (int dc = 0; dc < 32; ++dc) cacc[dc] = z;
  }

  for (int vt = 0; vt < NT; ++vt) {
    if (wave == 0) __builtin_amdgcn_s_wait_tensorcnt((short)0);
    __syncthreads();
    if (wave == 0 && vt + 1 < NT) kick(vt + 1, (vt + 1) & 1);
    convert_tile(vt & 1);
    __syncthreads();
    #pragma unroll
    for (int nsub = 0; nsub < BN / 16; ++nsub) {
      v8f c = computeS(nsub);
      #pragma unroll
      for (int v = 0; v < 8; ++v) {
        const float p = __expf(c[v] - mrun[v]) * linv[v];
        const int qrel = wave * 16 + grp * 8 + v;
        attn[((size_t)bb * LQ + qbase + qrel) * LV + (size_t)vt * BN + nsub * 16 + ln] = p;
        sP[wave][grp * 8 + v][nsub * 16 + ln] = f2bf(p);
      }
    }
    // intra-wave LDS RAW: P staging written above, read below as A-fragments
    asm volatile("s_wait_dscnt 0" ::: "memory");
    {
      FragBF aP;
      aP.q[0] = *reinterpret_cast<const uint4*>(&sP[wave][ln][grp * 8]);
      aP.q[1] = *reinterpret_cast<const uint4*>(&sP[wave][ln][grp * 8 + 16]);
      #pragma unroll
      for (int dc = 0; dc < 32; ++dc) {
        FragBF bC;
        const uint4* bp = reinterpret_cast<const uint4*>(&sVt[dc * 16 + ln][grp * 8]);
        bC.q[0] = bp[0]; bC.q[1] = bp[2];
        cacc[dc] = __builtin_amdgcn_wmma_f32_16x16x32_bf16(false, aP.v, false, bC.v,
                                                           (short)0, cacc[dc], false, false);
      }
    }
  }

  // ---- write context ----
  #pragma unroll
  for (int dc = 0; dc < 32; ++dc) {
    #pragma unroll
    for (int v = 0; v < 8; ++v) {
      const int qrel = wave * 16 + grp * 8 + v;
      ctx[((size_t)bb * LQ + qbase + qrel) * DIM + dc * 16 + ln] = cacc[dc][v];
    }
  }
}

extern "C" void kernel_launch(void* const* d_in, const int* in_sizes, int n_in,
                              void* d_out, int out_size, void* d_ws, size_t ws_size,
                              hipStream_t stream) {
  (void)in_sizes; (void)n_in; (void)out_size; (void)d_ws; (void)ws_size;
  const float* Q = (const float*)d_in[0];
  const float* V = (const float*)d_in[1];
  float* ctx  = (float*)d_out;                               // [B, Lq, D]
  float* attn = ctx + (size_t)NB * LQ * DIM;                 // [B, Lq, Lv]
  dim3 grid(NB * (LQ / BM));   // 16 batches * 16 q-tiles = 256 workgroups
  dim3 block(256);             // 8 waves (wave32)
  dpa_fused_kernel<<<grid, block, 0, stream>>>(Q, V, ctx, attn);
}
